// RNNModel_10041633538467
// MI455X (gfx1250) — compile-verified, matrix operand order checked by reading, not run on hardware
//
#include <hip/hip_runtime.h>
#include <hip/hip_bf16.h>

typedef __attribute__((ext_vector_type(16))) _Float16 v16h;
typedef __attribute__((ext_vector_type(8)))  float    v8f;

// Problem constants
#define BB   1024
#define TT   70
#define EE   300
#define HH   150
#define ZW4  600        // 4*H
#define NR   (BB*TT)    // 71680 rows for the input-projection GEMM
#define NTZ  38         // n-tiles for Wh (608/16)
#define NTX  40         // n-tiles for Wx pack (padded to 640 for 64-wide blocks)
#define KC5  5          // k-chunks of 32 covering K=150 (padded 160)
#define KCX  10         // k-chunks of 32 covering K=300 (padded 320)

// ---------------------------------------------------------------------------
// CDNA5 async global->LDS copy (ASYNCcnt) with safe fallback
// ---------------------------------------------------------------------------
#define GAS __attribute__((address_space(1)))
#define LAS __attribute__((address_space(3)))
#if defined(__has_builtin)
#if __has_builtin(__builtin_amdgcn_global_load_async_to_lds_b128) && \
    __has_builtin(__builtin_amdgcn_s_wait_asynccnt)
#define HAVE_ASYNC 1
#endif
#endif

typedef int v4i_ __attribute__((vector_size(16)));
typedef GAS v4i_* gasp_t;   // addrspace(1) int4*
typedef LAS v4i_* lasp_t;   // addrspace(3) int4*

__device__ __forceinline__ void copy16_async(void* lds, const void* g) {
#ifdef HAVE_ASYNC
    __builtin_amdgcn_global_load_async_to_lds_b128(
        (gasp_t)(unsigned long long)g, (lasp_t)lds, 0, 0);
#else
    *(float4*)lds = *(const float4*)g;
#endif
}

template <int N> __device__ __forceinline__ void wait_async() {
#ifdef HAVE_ASYNC
    __builtin_amdgcn_s_wait_asynccnt(N);
#endif
}

// ---------------------------------------------------------------------------
// sequence lengths from mask
// ---------------------------------------------------------------------------
__global__ void compute_lens(const int* __restrict__ mask, int* __restrict__ lens) {
    int b = blockIdx.x * blockDim.x + threadIdx.x;
    if (b >= BB) return;
    int s = 0;
    for (int t = 0; t < TT; ++t) s += mask[b * TT + t];
    lens[b] = s;
}

// ---------------------------------------------------------------------------
// Convert activations f32[NR,300] -> f16[NR,320] (zero padded K)
// ---------------------------------------------------------------------------
__global__ void cvt_f16(const float* __restrict__ in, _Float16* __restrict__ out16) {
    size_t idx = (size_t)blockIdx.x * blockDim.x + threadIdx.x;
    if (idx >= (size_t)NR * 320) return;
    int k = (int)(idx % 320);
    size_t row = idx / 320;
    out16[idx] = (_Float16)((k < EE) ? in[row * EE + k] : 0.f);
}

// ---------------------------------------------------------------------------
// Pack Wx (rows 0..299) into WMMA B-frag layout:
// [layer][dir][nt=40][kc=10][lane=32][16], n = nt*16+(lane&15),
// k = kc*32+(lane>>4)*16+j, zero padded.
// ---------------------------------------------------------------------------
__global__ void pack_wx(const float* __restrict__ fwk, const float* __restrict__ bwk,
                        _Float16* __restrict__ packed) {
    const int PER = NTX * KCX * 512;
    int idx = blockIdx.x * blockDim.x + threadIdx.x;
    if (idx >= 5 * 2 * PER) return;
    int ld = idx / PER, rem = idx % PER;
    int layer = ld >> 1, dir = ld & 1;
    int nt = rem / (KCX * 512); rem %= KCX * 512;
    int kc = rem / 512;         rem %= 512;
    int lane = rem >> 4, j = rem & 15;
    int n = nt * 16 + (lane & 15);
    int k = kc * 32 + (lane >> 4) * 16 + j;
    const float* kern = (dir ? bwk : fwk) + (size_t)layer * 450 * ZW4;
    float v = (n < ZW4 && k < EE) ? kern[(size_t)k * ZW4 + n] : 0.f;
    packed[idx] = (_Float16)v;
}

// ---------------------------------------------------------------------------
// Pack Wh (rows 300..449) into B-frag layout [layer][dir][nt=38][kc=5][512]
// ---------------------------------------------------------------------------
__global__ void pack_wh(const float* __restrict__ fwk, const float* __restrict__ bwk,
                        _Float16* __restrict__ packed) {
    const int PER = NTZ * KC5 * 512;
    int idx = blockIdx.x * blockDim.x + threadIdx.x;
    if (idx >= 5 * 2 * PER) return;
    int ld = idx / PER, rem = idx % PER;
    int layer = ld >> 1, dir = ld & 1;
    int nt = rem / (KC5 * 512); rem %= KC5 * 512;
    int kc = rem / 512;         rem %= 512;
    int lane = rem >> 4, j = rem & 15;
    int n = nt * 16 + (lane & 15);
    int k = kc * 32 + (lane >> 4) * 16 + j;
    const float* kern = (dir ? bwk : fwk) + (size_t)layer * 450 * ZW4;
    float v = (n < ZW4 && k < HH) ? kern[(size_t)(300 + k) * ZW4 + n] : 0.f;
    packed[idx] = (_Float16)v;
}

// ---------------------------------------------------------------------------
// Input projection GEMM: out[NR,600] = act16[NR,320] @ Wx + bias
// 128 threads / 4 waves; tile 64(M) x 64(N). A tiles async-DMA'd to LDS with
// double buffering (each wave stages only its own m-tile rows -> no barriers);
// B fragments come pre-packed from L2. blockIdx.z = direction.
// ---------------------------------------------------------------------------
__global__ __launch_bounds__(128) void xproj_gemm(
    const _Float16* __restrict__ act,   // [NR,320] f16
    const _Float16* __restrict__ pwx,   // this layer: [dir][40][10][512]
    const float* __restrict__ bfw, const float* __restrict__ bbw,
    float* __restrict__ outfw, float* __restrict__ outbw) {
    const int dir = blockIdx.z;
    const float* bias = dir ? bbw : bfw;
    float* out = dir ? outbw : outfw;
    const _Float16* pw = pwx + (size_t)dir * NTX * KCX * 512;
    const int m0 = blockIdx.x * 64;
    const int nt0 = blockIdx.y * 4;

    __shared__ __align__(32) _Float16 ldsA[2][64 * 32];

    const int tid = threadIdx.x;
    const int lane = tid & 31;
    const int wave = tid >> 5;

    // each lane stages half a row of its wave's m-tile
    const int arow = wave * 16 + (lane >> 1);
    const int ahalf = lane & 1;
    const _Float16* asrc = act + (size_t)(m0 + arow) * 320 + ahalf * 16;
    _Float16* adst0 = &ldsA[0][arow * 32 + ahalf * 16];
    _Float16* adst1 = &ldsA[1][arow * 32 + ahalf * 16];

    v8f acc[4];
    v8f vzero = {};
#pragma unroll
    for (int j = 0; j < 4; ++j) acc[j] = vzero;

    // prologue: stage chunk 0
    copy16_async(adst0, asrc);
    copy16_async(adst0 + 8, asrc + 8);

    for (int kc = 0; kc < KCX; ++kc) {
        const int buf = kc & 1;
        if (kc + 1 < KCX) {
            _Float16* d = (buf ? adst0 : adst1);
            const _Float16* s = asrc + (kc + 1) * 32;
            copy16_async(d, s);
            copy16_async(d + 8, s + 8);
            wait_async<2>();   // chunk kc complete; kc+1 may be in flight
        } else {
            wait_async<0>();
        }
        v16h a = *(const v16h*)(&ldsA[buf][(wave * 16 + (lane & 15)) * 32 + (lane >> 4) * 16]);
#pragma unroll
        for (int j = 0; j < 4; ++j) {
            const int nt = nt0 + j;
            v16h bf = *(const v16h*)(pw + ((size_t)nt * KCX + kc) * 512 + lane * 16);
            acc[j] = __builtin_amdgcn_wmma_f32_16x16x32_f16(
                false, a, false, bf, (short)0, acc[j], false, false);
        }
    }

    const int mbase = m0 + wave * 16 + (lane >> 4) * 8;
#pragma unroll
    for (int j = 0; j < 4; ++j) {
        int col = (nt0 + j) * 16 + (lane & 15);
        if (col < ZW4) {
            float bv = bias[col];
#pragma unroll
            for (int r = 0; r < 8; ++r)
                out[(size_t)(mbase + r) * ZW4 + col] = acc[j][r] + bv;
        }
    }
}

// ---------------------------------------------------------------------------
// Recurrent scan: workgroup = (16-row batch tile, direction), 256 threads.
// Per step: async-DMA the 16 Xproj rows (per-row time index; reversed for bw)
// into ldsZ, z += h @ Wh via WMMA (B-frags L2-resident), then TF LSTMCell
// gates (forget bias 1.0) with per-row masking; h carried f16 in LDS.
// ---------------------------------------------------------------------------
__global__ __launch_bounds__(256) void lstm_scan(
    const float* __restrict__ xproj_fw, const float* __restrict__ xproj_bw,
    const _Float16* __restrict__ packed,   // this layer: [dir][38][5][512]
    const int* __restrict__ lens,
    float* __restrict__ out)               // [B,T,300]: fw cols 0..149, bw 150..299
{
    const int b0  = blockIdx.x * 16;
    const int dir = blockIdx.y;
    const float* xp = dir ? xproj_bw : xproj_fw;
    const _Float16* pw = packed + (size_t)dir * NTZ * KC5 * 512;

    __shared__ __align__(32) _Float16 ldsH[16 * 160];
    __shared__ __align__(16) float ldsZ[16 * 608];
    __shared__ float ldsC[16 * 152];
    __shared__ int   ldsLen[16];
    __shared__ int   ldsT[16];
    __shared__ int   ldsAct[16];

    const int tid = threadIdx.x;
    const int lane = tid & 31;
    const int wave = tid >> 5;

    for (int i = tid; i < 16 * 160; i += 256) ldsH[i] = (_Float16)0.f;
    for (int i = tid; i < 16 * 152; i += 256) ldsC[i] = 0.f;
    if (tid < 16) ldsLen[tid] = lens[b0 + tid];
    __syncthreads();

    for (int s = 0; s < TT; ++s) {
        if (tid < 16) {
            int len = ldsLen[tid];
            int act = (s < len) ? 1 : 0;
            ldsAct[tid] = act;
            ldsT[tid] = (dir == 0) ? s : (act ? (len - 1 - s) : s);
        }
        __syncthreads();

        // async-stage z rows: 16 rows x 600 f32 (150 x b128 per row)
        for (int i = tid; i < 16 * 150; i += 256) {
            int row = i / 150;
            int c4 = i - row * 150;
            const float* g = xp + (((size_t)(b0 + row) * TT + ldsT[row]) * ZW4 + c4 * 4);
            copy16_async(&ldsZ[row * 608 + c4 * 4], g);
        }
        wait_async<0>();
        __syncthreads();

        const int mrow = (lane >> 4) * 8;
        const int ncol = lane & 15;
        for (int nt = wave; nt < NTZ; nt += 8) {
            int n = nt * 16 + ncol;
            v8f acc;
#pragma unroll
            for (int r = 0; r < 8; ++r)
                acc[r] = (n < ZW4) ? ldsZ[(mrow + r) * 608 + n] : 0.f;
            if (nt + 8 < NTZ)
                __builtin_prefetch(pw + ((size_t)(nt + 8) * KC5) * 512 + lane * 16, 0, 3);
#pragma unroll
            for (int kc = 0; kc < KC5; ++kc) {
                v16h a = *(const v16h*)(ldsH + (lane & 15) * 160 + kc * 32 + (lane >> 4) * 16);
                v16h bf = *(const v16h*)(pw + ((size_t)nt * KC5 + kc) * 512 + lane * 16);
                acc = __builtin_amdgcn_wmma_f32_16x16x32_f16(
                    false, a, false, bf, (short)0, acc, false, false);
            }
#pragma unroll
            for (int r = 0; r < 8; ++r)
                ldsZ[(mrow + r) * 608 + nt * 16 + ncol] = acc[r];
        }
        __syncthreads();

        // gates: TF order i, j, f, o ; forget bias 1.0
        for (int e = tid; e < 16 * HH; e += 256) {
            int row = e / HH, u = e - row * HH;
            float zi = ldsZ[row * 608 + u];
            float zj = ldsZ[row * 608 + u + 150];
            float zf = ldsZ[row * 608 + u + 300];
            float zo = ldsZ[row * 608 + u + 450];
            float c_old = ldsC[row * 152 + u];
            float fg = 1.f / (1.f + __expf(-(zf + 1.f)));
            float ig = 1.f / (1.f + __expf(-zi));
            float jt = tanhf(zj);
            float og = 1.f / (1.f + __expf(-zo));
            float c_new = c_old * fg + ig * jt;
            float h_new = tanhf(c_new) * og;
            int act = ldsAct[row];
            float c_keep = act ? c_new : c_old;
            float h_keep = act ? h_new : (float)ldsH[row * 160 + u];
            ldsC[row * 152 + u] = c_keep;
            ldsH[row * 160 + u] = (_Float16)h_keep;
            int t_out = act ? ldsT[row] : s;
            out[((size_t)(b0 + row) * TT + t_out) * 300 + dir * HH + u]
                = act ? h_new : 0.f;
        }
        __syncthreads();
    }
}

// ---------------------------------------------------------------------------
// Head: gather last valid step, dense+ReLU, logits, argmax.
// ---------------------------------------------------------------------------
__global__ __launch_bounds__(256) void head_kernel(
    const float* __restrict__ feat, const int* __restrict__ lens,
    const float* __restrict__ dW, const float* __restrict__ db,
    const float* __restrict__ W, const float* __restrict__ bvec,
    float* __restrict__ outv) {
    __shared__ float hrow[300];
    __shared__ float hlast[152];
    __shared__ float lg[2];
    int b = blockIdx.x;
    int idx = lens[b] - 1;
    if (idx < 0) idx = 0;
    const float* src = feat + ((size_t)b * TT + idx) * 300;
    for (int i = threadIdx.x; i < 300; i += 256) hrow[i] = src[i];
    __syncthreads();
    for (int u = threadIdx.x; u < HH; u += 256) {
        float a = db[u];
        for (int k = 0; k < 300; ++k) a += hrow[k] * dW[k * HH + u];
        hlast[u] = a > 0.f ? a : 0.f;
    }
    __syncthreads();
    if (threadIdx.x < 2) {
        int c = threadIdx.x;
        float a = bvec[c];
        for (int u = 0; u < HH; ++u) a += hlast[u] * W[u * 2 + c];
        lg[c] = a;
        outv[b * 2 + c] = a;
    }
    __syncthreads();
    if (threadIdx.x == 0)
        outv[2 * BB + b] = (lg[1] > lg[0]) ? 1.0f : 0.0f;
}

// ---------------------------------------------------------------------------
extern "C" void kernel_launch(void* const* d_in, const int* in_sizes, int n_in,
                              void* d_out, int out_size, void* d_ws, size_t ws_size,
                              hipStream_t stream) {
    const float* X    = (const float*)d_in[0];
    const int*   Xm   = (const int*)d_in[1];
    const float* fwk  = (const float*)d_in[2];
    const float* fwb  = (const float*)d_in[3];
    const float* bwk  = (const float*)d_in[4];
    const float* bwb  = (const float*)d_in[5];
    const float* dW   = (const float*)d_in[6];
    const float* db   = (const float*)d_in[7];
    const float* W    = (const float*)d_in[8];
    const float* bvec = (const float*)d_in[9];

    char* ws = (char*)d_ws;
    size_t off = 0;
    auto alloc = [&](size_t bytes) -> void* {
        void* p = ws + off;
        off += (bytes + 255) & ~(size_t)255;
        return p;
    };
    float*    xpf   = (float*)alloc((size_t)NR * ZW4 * 4);
    float*    xpb   = (float*)alloc((size_t)NR * ZW4 * 4);
    float*    bufA  = (float*)alloc((size_t)BB * TT * 300 * 4);
    float*    bufB  = (float*)alloc((size_t)BB * TT * 300 * 4);
    _Float16* act16 = (_Float16*)alloc((size_t)NR * 320 * 2);
    _Float16* pWh   = (_Float16*)alloc((size_t)10 * NTZ * KC5 * 512 * 2);
    _Float16* pWx   = (_Float16*)alloc((size_t)10 * NTX * KCX * 512 * 2);
    int*      lens  = (int*)alloc((size_t)BB * 4);

    compute_lens<<<(BB + 255) / 256, 256, 0, stream>>>(Xm, lens);
    {
        int total = 10 * NTZ * KC5 * 512;
        pack_wh<<<(total + 255) / 256, 256, 0, stream>>>(fwk, bwk, pWh);
    }
    {
        int total = 10 * NTX * KCX * 512;
        pack_wx<<<(total + 255) / 256, 256, 0, stream>>>(fwk, bwk, pWx);
    }

    const float* cur = X;
    float* nxt = bufA;
    for (int layer = 0; layer < 5; ++layer) {
        {
            size_t total = (size_t)NR * 320;
            cvt_f16<<<(unsigned)((total + 255) / 256), 256, 0, stream>>>(cur, act16);
        }
        const float* bf = fwb + (size_t)layer * ZW4;
        const float* bw = bwb + (size_t)layer * ZW4;
        const _Float16* px = pWx + (size_t)layer * 2 * NTX * KCX * 512;
        dim3 g(NR / 64, 10, 2);
        xproj_gemm<<<g, 128, 0, stream>>>(act16, px, bf, bw, xpf, xpb);
        const _Float16* ph = pWh + (size_t)layer * 2 * NTZ * KC5 * 512;
        dim3 gs(BB / 16, 2);
        lstm_scan<<<gs, 256, 0, stream>>>(xpf, xpb, ph, lens, nxt);
        cur = nxt;
        nxt = (nxt == bufA) ? bufB : bufA;
    }
    head_kernel<<<BB, 256, 0, stream>>>(cur, lens, dW, db, W, bvec, (float*)d_out);
}
